// RevolutionaryTransformerBlock_74363063763461
// MI455X (gfx1250) — compile-verified, matrix operand order checked by reading, not run on hardware
//
#include <hip/hip_runtime.h>
#include <hip/hip_bf16.h>
#include <math.h>

// ---------------------------------------------------------------------------
// RevolutionaryTransformerBlock for MI455X (gfx1250, wave32, WMMA).
// One tiled bf16-WMMA GEMM engine (fp32 -> bf16 RNE in the staging path,
// f32 accumulate) with fused epilogue, plus small elementwise kernels.
// v3: edge-tile staging is branchless (clamped address + select-zero) so
// boundary blocks compile to plain loads + v_cndmask instead of per-element
// exec-mask juggling; interior tiles keep the float4 -> bf16x4 fast path.
// Inner loop: 8 x v_wmma_f32_16x16x32_bf16 per wave per k-step.
// ---------------------------------------------------------------------------

static constexpr int kH  = 1024;   // hidden
static constexpr int kI  = 4096;   // intermediate
static constexpr int kE  = 8;      // experts
static constexpr int kK  = 4;      // top-k
static constexpr int kM  = 1000;   // memory slots
static constexpr int kNH = 8;      // heads
static constexpr int kDH = 128;    // head dim
static constexpr int kN  = 4096;   // B*S tokens

static constexpr int BM = 128, BN = 128, BK = 32;
static constexpr int LDP = 36;     // LDS row pitch in bf16 (72B: 8B-aligned rows)

typedef __attribute__((ext_vector_type(16))) __bf16 v16bf;
typedef __attribute__((ext_vector_type(4)))  __bf16 bf16x4;
typedef __attribute__((ext_vector_type(8)))  float  v8f;

__device__ __forceinline__ __bf16 f2bf(float f) {
  unsigned u = __builtin_bit_cast(unsigned, f);
  unsigned r = u + 0x7FFFu + ((u >> 16) & 1u);        // round-to-nearest-even
  unsigned short h = (unsigned short)(r >> 16);
  return __builtin_bit_cast(__bf16, h);
}

__device__ __forceinline__ float apply_act(float v, int act) {
  if (act == 1) return 0.5f * v * (1.0f + erff(v * 0.70710678118654752f)); // erf-GELU
  if (act == 2) return v / (1.0f + __expf(-v));                            // SiLU
  if (act == 3) return v > 0.f ? v : 0.f;                                  // ReLU
  return v;
}

// Branchless guarded load: clamp the address into range, select 0 if OOB.
__device__ __forceinline__ float guarded_ld(const float* __restrict__ P, int r, int rmax,
                                            int c, int cmax, int ld) {
  int rc = r < rmax ? r : rmax - 1;
  int cc = c < cmax ? c : cmax - 1;
  float v = P[(size_t)rc * ld + cc];
  return (r < rmax && c < cmax) ? v : 0.f;
}

// Assemble a 16x32 bf16 WMMA operand fragment from an LDS row (CDNA5 layout:
// lanes 0-15 hold K {0..7,16..23}, lanes 16-31 hold K {8..15,24..31}).
__device__ __forceinline__ v16bf load_frag(const __bf16* rowbase, int hi) {
  const bf16x4* p = (const bf16x4*)rowbase;            // 8B-aligned (pitch 72B)
  bf16x4 c0 = p[hi * 2 + 0];                           // K = hi*8   .. +3
  bf16x4 c1 = p[hi * 2 + 1];                           // K = hi*8+4 .. +7
  bf16x4 c2 = p[hi * 2 + 4];                           // K = 16+hi*8   .. +3
  bf16x4 c3 = p[hi * 2 + 5];                           // K = 16+hi*8+4 .. +7
  v16bf f;
  #pragma unroll
  for (int j = 0; j < 4; ++j) {
    f[j] = c0[j]; f[4 + j] = c1[j]; f[8 + j] = c2[j]; f[12 + j] = c3[j];
  }
  return f;
}

// ---------------------------------------------------------------------------
// Tiled WMMA GEMM:  C[n,d] = act(A[n,k] @ op(W) + bias + add) * mul
//   transB == 0 : W is [d,k] row-major  (torch Linear weight convention)
//   transB == 1 : W is [k,d] row-major  (att @ V with strided V)
// 256 threads = 8 wave32 in a 4(M) x 2(N) grid; each wave owns 32x64 =
// 2x4 tiles of v_wmma_f32_16x16x32_bf16.
// ---------------------------------------------------------------------------
__global__ __launch_bounds__(256)
void gemm_kernel(const float* __restrict__ A, int lda,
                 const float* __restrict__ W, int ldw, int transB,
                 const float* __restrict__ bias,
                 const float* __restrict__ addsrc,
                 const float* __restrict__ mulsrc,
                 float* __restrict__ C, int ldc,
                 int n, int k, int d, int act) {
  __shared__ __bf16 As[BM][LDP];
  __shared__ __bf16 Ws[BN][LDP];

  const int tid  = threadIdx.x;
  const int lane = tid & 31;
  const int wid  = tid >> 5;
  const int wm   = (wid >> 1) * 32;     // 0,32,64,96
  const int wn   = (wid & 1) * 64;      // 0,64
  const int l    = lane & 15;
  const int hi   = lane >> 4;

  const int bm0 = blockIdx.x * BM;
  const int bn0 = blockIdx.y * BN;

  const bool rowsA_full = (bm0 + BM <= n);
  const bool rowsW_full = (bn0 + BN <= d);

  v8f acc[2][4];
  #pragma unroll
  for (int i = 0; i < 2; ++i)
    #pragma unroll
    for (int j = 0; j < 4; ++j)
      acc[i][j] = v8f{0.f, 0.f, 0.f, 0.f, 0.f, 0.f, 0.f, 0.f};

  for (int k0 = 0; k0 < k; k0 += BK) {
    const bool kfull = (k0 + BK <= k);

    // ------------- stage A (BM x BK) fp32 -> bf16 -------------
    if (k0 + BK < k)  // prefetch next k-tile (global_prefetch_b8)
      __builtin_prefetch((const void*)(A + (size_t)(bm0 + (tid >> 1)) * lda + k0 + BK), 0, 1);
    if (rowsA_full && kfull) {
      // 4096 elems = 1024 float4 chunks; 4 chunks / thread, zero branches
      #pragma unroll
      for (int c = 0; c < 4; ++c) {
        int chunk = tid + c * 256;
        int r = chunk >> 3, col = (chunk & 7) * 4;
        const float4 v = *(const float4*)(A + (size_t)(bm0 + r) * lda + k0 + col);
        bf16x4 hh; hh[0] = f2bf(v.x); hh[1] = f2bf(v.y); hh[2] = f2bf(v.z); hh[3] = f2bf(v.w);
        *(bf16x4*)&As[r][col] = hh;
      }
    } else {
      #pragma unroll
      for (int i = 0; i < 16; ++i) {
        int e = tid + i * 256;
        int r = e >> 5, c = e & 31;
        As[r][c] = f2bf(guarded_ld(A, bm0 + r, n, k0 + c, k, lda));
      }
    }

    // ------------- stage W (BN x BK in [n][k] layout) -------------
    if (!transB) {
      if (rowsW_full && kfull) {
        #pragma unroll
        for (int c = 0; c < 4; ++c) {
          int chunk = tid + c * 256;
          int r = chunk >> 3, col = (chunk & 7) * 4;
          const float4 v = *(const float4*)(W + (size_t)(bn0 + r) * ldw + k0 + col);
          bf16x4 hh; hh[0] = f2bf(v.x); hh[1] = f2bf(v.y); hh[2] = f2bf(v.z); hh[3] = f2bf(v.w);
          *(bf16x4*)&Ws[r][col] = hh;
        }
      } else {
        #pragma unroll
        for (int i = 0; i < 16; ++i) {
          int e = tid + i * 256;
          int r = e >> 5, c = e & 31;
          Ws[r][c] = f2bf(guarded_ld(W, bn0 + r, d, k0 + c, k, ldw));
        }
      }
    } else {
      if (rowsW_full && kfull) {
        // tile is [BK=32][BN=128]; float4 along n, transpose into Ws[n][k]
        #pragma unroll
        for (int c = 0; c < 4; ++c) {
          int chunk = tid + c * 256;
          int kk = chunk >> 5, nn = (chunk & 31) * 4;
          const float4 v = *(const float4*)(W + (size_t)(k0 + kk) * ldw + bn0 + nn);
          Ws[nn + 0][kk] = f2bf(v.x);
          Ws[nn + 1][kk] = f2bf(v.y);
          Ws[nn + 2][kk] = f2bf(v.z);
          Ws[nn + 3][kk] = f2bf(v.w);
        }
      } else {
        #pragma unroll
        for (int i = 0; i < 16; ++i) {
          int e = tid + i * 256;
          int kk = e >> 7, nn = e & 127;
          Ws[nn][kk] = f2bf(guarded_ld(W, k0 + kk, k, bn0 + nn, d, ldw));
        }
      }
    }
    __syncthreads();

    // ------------- 8 x v_wmma_f32_16x16x32_bf16 per wave -------------
    v16bf af[2], bfm[4];
    #pragma unroll
    for (int i = 0; i < 2; ++i) af[i] = load_frag(&As[wm + i * 16 + l][0], hi);
    #pragma unroll
    for (int j = 0; j < 4; ++j) bfm[j] = load_frag(&Ws[wn + j * 16 + l][0], hi);
    #pragma unroll
    for (int i = 0; i < 2; ++i)
      #pragma unroll
      for (int j = 0; j < 4; ++j)
        acc[i][j] = __builtin_amdgcn_wmma_f32_16x16x32_bf16(
            false, af[i], false, bfm[j], (short)0, acc[i][j], false, false);
    __syncthreads();
  }

  // ------------- fused epilogue -------------
  #pragma unroll
  for (int i = 0; i < 2; ++i)
    #pragma unroll
    for (int j = 0; j < 4; ++j)
      #pragma unroll
      for (int r = 0; r < 8; ++r) {
        int row = bm0 + wm + i * 16 + hi * 8 + r;   // C/D: lanes 16-31 -> M+8
        int col = bn0 + wn + j * 16 + l;
        if (row < n && col < d) {
          float v = acc[i][j][r];
          if (bias)   v += bias[col];
          if (addsrc) v += addsrc[(size_t)row * ldc + col];
          v = apply_act(v, act);
          if (mulsrc) v *= mulsrc[(size_t)row * ldc + col];
          C[(size_t)row * ldc + col] = v;
        }
      }
}

// ---------------------------------------------------------------------------
// Router logits: one block/token; x staged in LDS; wave w computes expert w.
// ---------------------------------------------------------------------------
__global__ __launch_bounds__(256)
void router_kernel(const float* __restrict__ x, const float* __restrict__ rw,
                   const float* __restrict__ rb, const float* __restrict__ lb,
                   float* __restrict__ logits) {
  const int t = blockIdx.x, tid = threadIdx.x;
  __shared__ float xs[kH];
  #pragma unroll
  for (int i = 0; i < 4; ++i) xs[tid + i * 256] = x[(size_t)t * kH + tid + i * 256];
  __syncthreads();
  const int wid = tid >> 5, lane = tid & 31;
  float acc = 0.f;
  for (int j = lane; j < kH; j += 32) acc += xs[j] * rw[(size_t)wid * kH + j];
  #pragma unroll
  for (int off = 16; off > 0; off >>= 1) acc += __shfl_down(acc, off, 32);
  if (lane == 0) logits[(size_t)t * kE + wid] = acc + rb[wid] + lb[wid];
}

__global__ void topk_kernel(const float* __restrict__ logits,
                            float* __restrict__ topw, int* __restrict__ topi) {
  int t = blockIdx.x * blockDim.x + threadIdx.x;
  if (t >= kN) return;
  float lg[kE];
  #pragma unroll
  for (int e = 0; e < kE; ++e) lg[e] = logits[(size_t)t * kE + e];
  float wv[kK]; int wi[kK]; bool used[kE] = {};
  #pragma unroll
  for (int kk = 0; kk < kK; ++kk) {
    float best = -3.4e38f; int bi = 0;
    #pragma unroll
    for (int e = 0; e < kE; ++e)
      if (!used[e] && lg[e] > best) { best = lg[e]; bi = e; }
    used[bi] = true; wv[kk] = best; wi[kk] = bi;
  }
  float mx = wv[0], sum = 0.f, ev[kK];
  #pragma unroll
  for (int kk = 0; kk < kK; ++kk) { ev[kk] = __expf(wv[kk] - mx); sum += ev[kk]; }
  float inv = 1.f / sum;
  #pragma unroll
  for (int kk = 0; kk < kK; ++kk) {
    topw[(size_t)t * kK + kk] = ev[kk] * inv;
    topi[(size_t)t * kK + kk] = wi[kk];
  }
}

// Residual add + LayerNorm over H, one block/token.
__global__ __launch_bounds__(256)
void add_layernorm_kernel(const float* __restrict__ a, const float* __restrict__ b,
                          const float* __restrict__ g, const float* __restrict__ bt,
                          float* __restrict__ out) {
  const int t = blockIdx.x, tid = threadIdx.x;
  __shared__ float red[256];
  float vals[4]; float s = 0.f;
  #pragma unroll
  for (int i = 0; i < 4; ++i) {
    int j = tid + i * 256;
    float v = a[(size_t)t * kH + j] + b[(size_t)t * kH + j];
    vals[i] = v; s += v;
  }
  red[tid] = s; __syncthreads();
  for (int o = 128; o > 0; o >>= 1) { if (tid < o) red[tid] += red[tid + o]; __syncthreads(); }
  float mean = red[0] / (float)kH; __syncthreads();
  float vs = 0.f;
  #pragma unroll
  for (int i = 0; i < 4; ++i) { float d0 = vals[i] - mean; vs += d0 * d0; }
  red[tid] = vs; __syncthreads();
  for (int o = 128; o > 0; o >>= 1) { if (tid < o) red[tid] += red[tid + o]; __syncthreads(); }
  float inv = rsqrtf(red[0] / (float)kH + 1e-5f);
  #pragma unroll
  for (int i = 0; i < 4; ++i) {
    int j = tid + i * 256;
    out[(size_t)t * kH + j] = (vals[i] - mean) * inv * g[j] + bt[j];
  }
}

// Row softmax with scale, valid length m, row stride ld. One block/row.
__global__ __launch_bounds__(256)
void softmax_rows_kernel(float* __restrict__ s, int ld, int m, float scale) {
  const int t = blockIdx.x, tid = threadIdx.x;
  __shared__ float red[256];
  float vals[4]; float mx = -3.4e38f;
  #pragma unroll
  for (int i = 0; i < 4; ++i) {
    int j = tid + i * 256;
    float v = (j < m) ? s[(size_t)t * ld + j] * scale : -3.4e38f;
    vals[i] = v; mx = fmaxf(mx, v);
  }
  red[tid] = mx; __syncthreads();
  for (int o = 128; o > 0; o >>= 1) { if (tid < o) red[tid] = fmaxf(red[tid], red[tid + o]); __syncthreads(); }
  mx = red[0]; __syncthreads();
  float sum = 0.f;
  #pragma unroll
  for (int i = 0; i < 4; ++i) {
    int j = tid + i * 256;
    if (j < m) { vals[i] = __expf(vals[i] - mx); sum += vals[i]; }
  }
  red[tid] = sum; __syncthreads();
  for (int o = 128; o > 0; o >>= 1) { if (tid < o) red[tid] += red[tid + o]; __syncthreads(); }
  float inv = 1.f / red[0];
  #pragma unroll
  for (int i = 0; i < 4; ++i) {
    int j = tid + i * 256;
    if (j < m) s[(size_t)t * ld + j] = vals[i] * inv;
  }
}

// out[t,:] += w_e(t) * eout[t,:] for tokens that routed to expert e.
__global__ __launch_bounds__(256)
void accum_kernel(const float* __restrict__ eout, const int* __restrict__ topi,
                  const float* __restrict__ topw, int e, float* __restrict__ out) {
  const int t = blockIdx.x, tid = threadIdx.x;
  float w = 0.f;
  #pragma unroll
  for (int kk = 0; kk < kK; ++kk)
    if (topi[(size_t)t * kK + kk] == e) w = topw[(size_t)t * kK + kk];
  if (w == 0.f) return;
  #pragma unroll
  for (int i = 0; i < 4; ++i) {
    int j = tid + i * 256;
    out[(size_t)t * kH + j] += w * eout[(size_t)t * kH + j];
  }
}

// ---------------------------------------------------------------------------
extern "C" void kernel_launch(void* const* d_in, const int* in_sizes, int n_in,
                              void* d_out, int out_size, void* d_ws, size_t ws_size,
                              hipStream_t stream) {
  (void)in_sizes; (void)n_in; (void)ws_size;
  auto in = [&](int i) { return (const float*)d_in[i]; };
  const float* x = in(0);
  float* out = (float*)d_out;

  // Input index map (setup_inputs dict order, depth-first):
  // 0:x 1:router_w 2:router_b 3:load_balancer
  // sw0 4:w1 5:w3 6:w2 | sw7 7:w1 8:w3 9:w2
  // math 10:eq_w 11:eq_b 12:v_w 13:v_b 14:o_w 15:o_b 16:c1_w 17:c1_b 18:c2_w 19:c2_b
  // code 20:syn_w 21:syn_b 22:v_w 23:v_b 24:o_w 25:o_b 26:ln1_g 27:ln1_b
  //      28:ff1_w 29:ff1_b 30:ff2_w 31:ff2_b 32:ln2_g 33:ln2_b 34:gen_w 35:gen_b
  // ffn 36:w1 37:b1 38:w2 39:b2
  // mem 40:enc_w 41:enc_b 42:q_w 43:q_b 44:k_w 45:k_b 46:v_w 47:v_b 48:o_w 49:o_b
  //     50:bank 51:con_w 52:con_b
  // vis 53..60 | aud 61..68 (e1_w,e1_b,e2_w,e2_b,v_w,v_b,o_w,o_b)

  // ---- workspace layout (floats), ~160MB ----
  float* wsp = (float*)d_ws;
  size_t off = 0;
  auto alloc = [&](size_t nf) { float* p = wsp + off; off += nf; return p; };
  float* b_logits = alloc((size_t)kN * kE);
  float* b_topw   = alloc((size_t)kN * kK);
  int*   b_topi   = (int*)alloc((size_t)kN * kK);
  float* b_big    = alloc((size_t)kN * kI);        // [N,I] intermediate
  float* b_h1     = alloc((size_t)kN * kH);
  float* b_h2     = alloc((size_t)kN * kH);
  float* b_h3     = alloc((size_t)kN * kH);
  float* b_eout   = alloc((size_t)kN * kH);        // current expert output
  float* b_scores = alloc((size_t)kN * 1024);      // [N,1024], valid cols < M
  float* b_kbank  = alloc((size_t)1024 * kH);
  float* b_vbank  = alloc((size_t)1024 * kH);

  auto gemm = [&](const float* A, int lda, const float* W, int ldw, int tb,
                  const float* bias, const float* add, const float* mul,
                  float* C, int ldc, int n, int k, int d, int act) {
    dim3 g((n + BM - 1) / BM, (d + BN - 1) / BN);
    gemm_kernel<<<g, dim3(256), 0, stream>>>(A, lda, W, ldw, tb, bias, add, mul,
                                             C, ldc, n, k, d, act);
  };
  auto accum = [&](int e) {
    accum_kernel<<<kN, 256, 0, stream>>>(b_eout, b_topi, b_topw, e, out);
  };

  // ---- routing ----
  router_kernel<<<kN, 256, 0, stream>>>(x, in(1), in(2), in(3), b_logits);
  topk_kernel<<<(kN + 255) / 256, 256, 0, stream>>>(b_logits, b_topw, b_topi);
  hipMemsetAsync(d_out, 0, (size_t)out_size * sizeof(float), stream);

  // ---- experts 0 & 7: SwiGLU ----
  auto swiglu = [&](const float* w1, const float* w3, const float* w2, int e) {
    gemm(x, kH, w3, kH, 0, nullptr, nullptr, nullptr, b_big, kI, kN, kH, kI, 0);     // t3
    gemm(x, kH, w1, kH, 0, nullptr, nullptr, b_big,  b_big, kI, kN, kH, kI, 2);      // silu(xw1)*t3
    gemm(b_big, kI, w2, kI, 0, nullptr, nullptr, nullptr, b_eout, kH, kN, kI, kH, 0);
    accum(e);
  };
  swiglu(in(4), in(5), in(6), 0);

  // ---- expert 1: math ----
  gemm(x,     kH, in(10), kH,   0, in(11), 0, 0, b_h1,  kH,   kN, kH,   kH,   0); // eq
  gemm(b_h1,  kH, in(12), kH,   0, in(13), 0, 0, b_h2,  kH,   kN, kH,   kH,   0); // v
  gemm(b_h2,  kH, in(14), kH,   0, in(15), 0, 0, b_h3,  kH,   kN, kH,   kH,   0); // sym
  gemm(b_h3,  kH, in(16), kH,   0, in(17), 0, 0, b_big, 2*kH, kN, kH,   2*kH, 1); // gelu(c1)
  gemm(b_big, 2*kH, in(18), 2*kH, 0, in(19), 0, 0, b_eout, kH, kN, 2*kH, kH,  0); // c2
  accum(1);

  // ---- expert 2: code (post-norm transformer layer) ----
  gemm(x,    kH, in(20), kH, 0, in(21), 0, 0, b_h1, kH, kN, kH, kH, 0);           // syn
  gemm(b_h1, kH, in(22), kH, 0, in(23), 0, 0, b_h2, kH, kN, kH, kH, 0);           // v
  gemm(b_h2, kH, in(24), kH, 0, in(25), 0, 0, b_h3, kH, kN, kH, kH, 0);           // attn out
  add_layernorm_kernel<<<kN, 256, 0, stream>>>(b_h1, b_h3, in(26), in(27), b_h2); // h1
  gemm(b_h2,  kH,   in(28), kH,   0, in(29), 0, 0, b_big, 2*kH, kN, kH,   2*kH, 3); // relu ff1
  gemm(b_big, 2*kH, in(30), 2*kH, 0, in(31), 0, 0, b_h3,  kH,   kN, 2*kH, kH,   0); // ff2
  add_layernorm_kernel<<<kN, 256, 0, stream>>>(b_h2, b_h3, in(32), in(33), b_h1); // h2
  gemm(b_h1, kH, in(34), kH, 0, in(35), 0, 0, b_eout, kH, kN, kH, kH, 0);         // gen
  accum(2);

  // ---- expert 3: ffn ----
  gemm(x,     kH, in(36), kH, 0, in(37), 0, 0, b_big,  kI, kN, kH, kI, 1);        // gelu
  gemm(b_big, kI, in(38), kI, 0, in(39), 0, 0, b_eout, kH, kN, kI, kH, 0);
  accum(3);

  // ---- expert 4: memory-bank attention ----
  gemm(x,      kH, in(40), kH, 0, in(41), 0, 0, b_h1,    kH, kN, kH, kH, 0);      // enc
  gemm(b_h1,   kH, in(42), kH, 0, in(43), 0, 0, b_h2,    kH, kN, kH, kH, 0);      // q
  gemm(in(50), kH, in(44), kH, 0, in(45), 0, 0, b_kbank, kH, kM, kH, kH, 0);      // K(bank)
  gemm(in(50), kH, in(46), kH, 0, in(47), 0, 0, b_vbank, kH, kM, kH, kH, 0);      // V(bank)
  for (int h = 0; h < kNH; ++h) {
    gemm(b_h2 + h * kDH, kH, b_kbank + h * kDH, kH, 0, nullptr, nullptr, nullptr,
         b_scores, 1024, kN, kDH, kM, 0);                                         // Q_h K_h^T
    softmax_rows_kernel<<<kN, 256, 0, stream>>>(b_scores, 1024, kM,
                                                0.08838834764831845f);            // 1/sqrt(128)
    gemm(b_scores, 1024, b_vbank + h * kDH, kH, 1, nullptr, nullptr, nullptr,
         b_h3 + h * kDH, kH, kN, kM, kDH, 0);                                     // att V_h
  }
  gemm(b_h3, kH, in(48), kH, 0, in(49), b_h1, 0, b_h2,   kH, kN, kH, kH, 0);      // o-proj + enc
  gemm(b_h2, kH, in(51), kH, 0, in(52), 0,    0, b_eout, kH, kN, kH, kH, 0);      // con-proj
  accum(4);

  // ---- experts 5 & 6: vis / aud ----
  auto va = [&](int base, int e) {
    gemm(x,    kH, in(base + 0), kH, 0, in(base + 1), 0, 0, b_h1,   kH, kN, kH, kH, 1);
    gemm(b_h1, kH, in(base + 2), kH, 0, in(base + 3), 0, 0, b_h2,   kH, kN, kH, kH, 0);
    gemm(b_h2, kH, in(base + 4), kH, 0, in(base + 5), 0, 0, b_h3,   kH, kN, kH, kH, 0);
    gemm(b_h3, kH, in(base + 6), kH, 0, in(base + 7), 0, 0, b_eout, kH, kN, kH, kH, 0);
    accum(e);
  };
  va(53, 5);
  va(61, 6);

  swiglu(in(7), in(8), in(9), 7);
}
